// SegFormerX_2173253452002
// MI455X (gfx1250) — compile-verified, matrix operand order checked by reading, not run on hardware
//
#include <hip/hip_runtime.h>

// ---------------------------------------------------------------------------
// Types
// ---------------------------------------------------------------------------
typedef __attribute__((ext_vector_type(16))) __bf16 v16bf;
typedef __attribute__((ext_vector_type(2)))  __bf16 v2bf;
typedef __attribute__((ext_vector_type(8)))  float  v8f;
typedef __attribute__((ext_vector_type(4)))  unsigned int v4u;

union FragU { v16bf v; v4u q[2]; };

// Load a 16x32 bf16 fragment (A-operand layout; B mirrors it) from row-major
// memory: lane l (l<16) holds row l, K quarters {0-7,16-23}; lane l+16 holds
// row l, K quarters {8-15,24-31}. Matches CDNA5 ISA 16-bit A layout.
__device__ __forceinline__ v16bf ld_frag(const __bf16* base, int stride) {
  const int lane = threadIdx.x & 31;
  const int r    = lane & 15;
  const int half = lane >> 4;
  const __bf16* p = base + (size_t)r * stride + half * 8;
  FragU f;
  f.q[0] = *(const v4u*)(p);        // K quarter 0 (or 1)
  f.q[1] = *(const v4u*)(p + 16);   // K quarter 2 (or 3)
  return f.v;
}

__device__ __forceinline__ v8f wmma_bf16(v16bf a, v16bf b, v8f c) {
  return __builtin_amdgcn_wmma_f32_16x16x32_bf16(false, a, false, b,
                                                 (short)0, c, false, false);
}

__device__ __forceinline__ float fexp2(float x) {
#if __has_builtin(__builtin_amdgcn_exp2f)
  return __builtin_amdgcn_exp2f(x);
#else
  return exp2f(x);
#endif
}

// gfx1250 async copy: 16B global -> LDS, tracked by ASYNCcnt.
// ldsAddr is the 32-bit LDS byte address (low 32 bits of the generic pointer
// to a __shared__ object), gaddr the 64-bit global address (GV mode).
__device__ __forceinline__ void async_b128_to_lds(unsigned ldsAddr,
                                                  const void* gaddr) {
  asm volatile("global_load_async_to_lds_b128 %0, %1, off"
               :
               : "v"(ldsAddr), "v"(gaddr)
               : "memory");
}

__device__ __forceinline__ void wait_async0() {
#if __has_builtin(__builtin_amdgcn_s_wait_asynccnt)
  __builtin_amdgcn_s_wait_asynccnt(0);
#else
  asm volatile("s_wait_asynccnt 0x0" ::: "memory");
#endif
}

// ---------------------------------------------------------------------------
// Elementwise conversions
// ---------------------------------------------------------------------------
__global__ void cvt_bf16_k(const float* __restrict__ x, __bf16* __restrict__ y,
                           int n) {
  int i = blockIdx.x * 256 + threadIdx.x;
  if (i < n) y[i] = (__bf16)x[i];
}

// W[512,512] row-major (k-major) -> Wt[n,k] bf16 (so B-fragments are
// contiguous row loads).
__global__ void transpose_cvt_k(const float* __restrict__ W,
                                __bf16* __restrict__ Wt) {
  int idx = blockIdx.x * 256 + threadIdx.x;   // 0 .. 512*512-1
  int n = idx >> 9, k = idx & 511;
  Wt[idx] = (__bf16)W[k * 512 + n];
}

// ---------------------------------------------------------------------------
// Projection / FFN GEMM:  C[M,512] = A_bf16[M,512] @ W[512,512] + bias
// block = 256 threads = 8 waves; wave w owns 64 output columns; each block
// owns 16 rows.  The block's A-tile (16x512 bf16 = 16 KB, contiguous) is
// staged once into LDS via gfx1250 async-to-LDS copies and shared by all 8
// waves (8x reduction in global A traffic).  Output modes:
//   0: bf16 [B, H, Ltot, 64]     (Q / K per-head layout)
//   1: bf16 [B, H, 64, Ltot]     (V transposed, col_off for concat position)
//   2: f32  [B, M, 512]          (FFN output, pre-LayerNorm)
// ---------------------------------------------------------------------------
__global__ __launch_bounds__(256)
void gemm_bf16_k(const __bf16* __restrict__ A, const __bf16* __restrict__ Wt,
                 const float* __restrict__ bias, void* __restrict__ out,
                 int M, int mode, int Ltot, int colOff) {
  __shared__ __align__(16) __bf16 Atile[16 * 512];   // 16 KB

  const int b    = blockIdx.z;
  const int wid  = threadIdx.x >> 5;
  const int lane = threadIdx.x & 31;
  const int ln   = lane & 15;
  const int half = lane >> 4;
  const int m0 = blockIdx.x * 16;
  const int n0 = wid * 64;

  // ---- stage A tile into LDS (4 x b128 async copies per thread) ---------
  {
    const char* gsrc = (const char*)(A + ((size_t)b * M + m0) * 512);
    unsigned ldsBase = (unsigned)(size_t)(&Atile[0]);
    int t16 = threadIdx.x * 16;
#pragma unroll
    for (int i = 0; i < 4; i++) {
      int byteOff = t16 + i * 4096;
      async_b128_to_lds(ldsBase + byteOff, gsrc + byteOff);
    }
    wait_async0();
    __syncthreads();
  }

  v8f acc[4] = {};
  for (int k = 0; k < 512; k += 32) {
    v16bf af = ld_frag(Atile + k, 512);   // ds_load_b128 from LDS
#pragma unroll
    for (int t = 0; t < 4; t++) {
      v16bf bf_ = ld_frag(Wt + (size_t)(n0 + t * 16) * 512 + k, 512);
      acc[t] = wmma_bf16(af, bf_, acc[t]);
    }
  }

  // C layout: vgpr r -> row m0 + r + 8*half ; col = n0 + t*16 + ln
  if (mode == 0) {
    __bf16* o = (__bf16*)out;  // [B, 8, Ltot, 64]; head = wid, d = t*16+ln
#pragma unroll
    for (int t = 0; t < 4; t++) {
      int d = t * 16 + ln;
      float bv = bias[n0 + d];
      size_t base = (((size_t)b * 8 + wid) * Ltot + m0 + 8 * half) * 64 + d;
#pragma unroll
      for (int r = 0; r < 8; r++)
        o[base + (size_t)r * 64] = (__bf16)(acc[t][r] + bv);
    }
  } else if (mode == 1) {
    __bf16* o = (__bf16*)out;  // [B, 8, 64, Ltot]
#pragma unroll
    for (int t = 0; t < 4; t++) {
      int d = t * 16 + ln;
      float bv = bias[n0 + d];
      size_t base =
          (((size_t)b * 8 + wid) * 64 + d) * (size_t)Ltot + colOff + m0 + 8 * half;
#pragma unroll
      for (int r = 0; r < 8; r += 2) {
        v2bf pr;
        pr[0] = (__bf16)(acc[t][r] + bv);
        pr[1] = (__bf16)(acc[t][r + 1] + bv);
        *(v2bf*)(o + base + r) = pr;
      }
    }
  } else {
    float* o = (float*)out;  // [B, M, 512]
#pragma unroll
    for (int t = 0; t < 4; t++) {
      int n = n0 + t * 16 + ln;
      float bv = bias[n];
#pragma unroll
      for (int r = 0; r < 8; r++) {
        int m = m0 + r + 8 * half;
        o[((size_t)b * M + m) * 512 + n] = acc[t][r] + bv;
      }
    }
  }
}

// ---------------------------------------------------------------------------
// Fused flash attention (transposed formulation), 1 wave per 16 queries.
// Keys 0..Lka-1 use (Qa,Ka,mka); keys Lka.. use (Qb,Kb,mkb).
// Vt: [B, 8, 64, Lka+Lkb] bf16 (values pre-transposed). Out: [B, Lq, 512] bf16.
// Mask semantics match reference: where(mq&mk, s, -10000) then * 1/sqrt(64).
// ---------------------------------------------------------------------------
__global__ __launch_bounds__(32)
void attn_fused_k(const __bf16* __restrict__ Qa, const __bf16* __restrict__ Qb,
                  const __bf16* __restrict__ Ka, const __bf16* __restrict__ Kb,
                  const __bf16* __restrict__ Vt,
                  const unsigned char* __restrict__ mq,
                  const unsigned char* __restrict__ mka,
                  const unsigned char* __restrict__ mkb,
                  __bf16* __restrict__ outp, int Lq, int Lka, int Lkb) {
  const float L2E   = 1.44269504f;
  const float SCALE = 0.125f;    // 1/sqrt(64)
  const float NEGS  = -1250.0f;  // -10000 * SCALE

  const int lane = threadIdx.x & 31;
  const int ln   = lane & 15;
  const int half = lane >> 4;
  const int q0 = blockIdx.x * 16;
  const int h  = blockIdx.y;
  const int b  = blockIdx.z;
  const int Kt = Lka + Lkb;

  const int  qIdx = q0 + ln;                      // this lane's query column
  const bool qm   = mq[(size_t)b * Lq + qIdx] != 0;

  const __bf16* qaB = Qa + (((size_t)b * 8 + h) * Lq + q0) * 64;
  const __bf16* qbB = Qb + (((size_t)b * 8 + h) * Lq + q0) * 64;
  v16bf qa0 = ld_frag(qaB, 64),      qa1 = ld_frag(qaB + 32, 64);
  v16bf qb0 = ld_frag(qbB, 64),      qb1 = ld_frag(qbB + 32, 64);

  v8f acc[4] = {};                 // O^T: 4 d-tiles x (16d x 16q)
  float mrun = -3.0e38f, lrun = 0.0f;

  for (int kb = 0; kb < Kt; kb += 32) {
    const bool inA = kb < Lka;
    const __bf16* kbase =
        inA ? (Ka + (((size_t)b * 8 + h) * Lka + kb) * 64)
            : (Kb + (((size_t)b * 8 + h) * Lkb + (kb - Lka)) * 64);
    const unsigned char* mkp = inA ? (mka + (size_t)b * Lka + kb)
                                   : (mkb + (size_t)b * Lkb + (kb - Lka));
    v16bf qf0 = inA ? qa0 : qb0;
    v16bf qf1 = inA ? qa1 : qb1;

    // S^T tiles: rows = keys, cols = queries, chained over DH=64 (2 chunks)
    v8f c0 = {}, c1 = {};
    c0 = wmma_bf16(ld_frag(kbase, 64),            qf0, c0);
    c0 = wmma_bf16(ld_frag(kbase + 32, 64),       qf1, c0);
    c1 = wmma_bf16(ld_frag(kbase + 16 * 64, 64),      qf0, c1);
    c1 = wmma_bf16(ld_frag(kbase + 16 * 64 + 32, 64), qf1, c1);

    // mask + scale; C row r in this lane-half is key kb + sub*16 + 8*half + r
    float s0[8], s1[8];
    float bm = -3.0e38f;
#pragma unroll
    for (int r = 0; r < 8; r++) {
      bool k0 = mkp[8 * half + r] != 0;
      bool k1 = mkp[16 + 8 * half + r] != 0;
      s0[r] = (qm && k0) ? c0[r] * SCALE : NEGS;
      s1[r] = (qm && k1) ? c1[r] * SCALE : NEGS;
      bm = fmaxf(bm, fmaxf(s0[r], s1[r]));
    }
    bm = fmaxf(bm, __shfl_xor(bm, 16));   // combine the two key-halves
    float mnew  = fmaxf(mrun, bm);
    float alpha = fexp2((mrun - mnew) * L2E);

    // exponentiate and repack directly into the B-operand fragment (P^T):
    // C rows of tile0 -> K quarters 0/1, tile1 -> quarters 2/3 (no shuffles)
    float ls = 0.0f;
    v16bf pf;
#pragma unroll
    for (int r = 0; r < 8; r++) {
      float p0 = fexp2((s0[r] - mnew) * L2E);
      float p1 = fexp2((s1[r] - mnew) * L2E);
      ls += p0 + p1;
      pf[r]     = (__bf16)p0;
      pf[8 + r] = (__bf16)p1;
    }
    ls += __shfl_xor(ls, 16);
    lrun = lrun * alpha + ls;
    mrun = mnew;

#pragma unroll
    for (int t = 0; t < 4; t++)
#pragma unroll
      for (int r = 0; r < 8; r++) acc[t][r] *= alpha;

    // O^T += V^T_tile (16d x 32k, contiguous rows of Vt) x P^T
    const __bf16* vb = Vt + (((size_t)b * 8 + h) * 64) * (size_t)Kt + kb;
#pragma unroll
    for (int t = 0; t < 4; t++) {
      v16bf vf = ld_frag(vb + (size_t)(t * 16) * Kt, Kt);
      acc[t] = wmma_bf16(vf, pf, acc[t]);
    }
  }

  // epilogue: out[b, q, h*64 + d] = O^T[d,q] / l[q]   (d = t*16 + r + 8*half)
  float inv = 1.0f / lrun;
  __bf16* op = outp + ((size_t)b * Lq + qIdx) * 512 + h * 64 + 8 * half;
#pragma unroll
  for (int t = 0; t < 4; t++) {
#pragma unroll
    for (int r = 0; r < 8; r += 2) {
      v2bf pr;
      pr[0] = (__bf16)(acc[t][r] * inv);
      pr[1] = (__bf16)(acc[t][r + 1] * inv);
      *(v2bf*)(op + t * 16 + r) = pr;
    }
  }
}

// ---------------------------------------------------------------------------
// Residual + LayerNorm over D=512: out = LN(feat + y) * g + b
// ---------------------------------------------------------------------------
__global__ __launch_bounds__(256)
void add_ln_k(const float* __restrict__ feat, const float* __restrict__ y,
              const float* __restrict__ g, const float* __restrict__ be,
              float* __restrict__ out) {
  const int row = blockIdx.x;
  const int t = threadIdx.x;
  const size_t base = (size_t)row * 512;
  float x0 = feat[base + t] + y[base + t];
  float x1 = feat[base + t + 256] + y[base + t + 256];
  float s = x0 + x1, q = x0 * x0 + x1 * x1;
#pragma unroll
  for (int o = 1; o < 32; o <<= 1) {
    s += __shfl_xor(s, o);
    q += __shfl_xor(q, o);
  }
  __shared__ float shs[8], shq[8];
  const int w = t >> 5;
  if ((t & 31) == 0) { shs[w] = s; shq[w] = q; }
  __syncthreads();
  float S = 0.0f, Q = 0.0f;
#pragma unroll
  for (int i = 0; i < 8; i++) { S += shs[i]; Q += shq[i]; }
  float mu   = S * (1.0f / 512.0f);
  float var  = Q * (1.0f / 512.0f) - mu * mu;
  float rstd = rsqrtf(var + 1e-12f);
  out[base + t]       = (x0 - mu) * rstd * g[t] + be[t];
  out[base + t + 256] = (x1 - mu) * rstd * g[t + 256] + be[t + 256];
}

// ---------------------------------------------------------------------------
// Host launcher
// ---------------------------------------------------------------------------
extern "C" void kernel_launch(void* const* d_in, const int* in_sizes, int n_in,
                              void* d_out, int out_size, void* d_ws,
                              size_t ws_size, hipStream_t stream) {
  (void)in_sizes; (void)n_in; (void)out_size; (void)ws_size;
  constexpr int Bc = 2, LV = 2048, LT = 512, Dc = 512;
  constexpr int KT = LV + LT;      // 2560 concatenated keys
  constexpr size_t DD = (size_t)Dc * Dc;

  const float* vid_feat = (const float*)d_in[0];
  const unsigned char* vmask = (const unsigned char*)d_in[1];
  const float* usr_feat = (const float*)d_in[2];
  const unsigned char* umask = (const unsigned char*)d_in[3];
  const float* Wq = (const float*)d_in[4];
  const float* bq = (const float*)d_in[5];
  const float* Wk = (const float*)d_in[6];
  const float* bk = (const float*)d_in[7];
  const float* Wv = (const float*)d_in[8];
  const float* bv = (const float*)d_in[9];
  const float* ffvW = (const float*)d_in[10];
  const float* ffvb = (const float*)d_in[11];
  const float* ffuW = (const float*)d_in[12];
  const float* ffub = (const float*)d_in[13];
  const float* lnvg = (const float*)d_in[14];
  const float* lnvb = (const float*)d_in[15];
  const float* lnug = (const float*)d_in[16];
  const float* lnub = (const float*)d_in[17];

  // ---- carve workspace -------------------------------------------------
  size_t off = 0;
  auto carve = [&](size_t bytes) -> void* {
    void* p = (char*)d_ws + off;
    off += (bytes + 255) & ~(size_t)255;
    return p;
  };
  __bf16* vid_bf = (__bf16*)carve((size_t)Bc * LV * Dc * 2);
  __bf16* usr_bf = (__bf16*)carve((size_t)Bc * LT * Dc * 2);
  __bf16* Wbf    = (__bf16*)carve(14 * DD * 2);  // Wq0-3,Wk0-3,Wv0-3,ffv,ffu
  __bf16* Q0 = (__bf16*)carve((size_t)Bc * LV * Dc * 2);
  __bf16* Q1 = (__bf16*)carve((size_t)Bc * LV * Dc * 2);
  __bf16* Q2 = (__bf16*)carve((size_t)Bc * LT * Dc * 2);
  __bf16* Q3 = (__bf16*)carve((size_t)Bc * LT * Dc * 2);
  __bf16* K0 = (__bf16*)carve((size_t)Bc * LV * Dc * 2);
  __bf16* K1 = (__bf16*)carve((size_t)Bc * LT * Dc * 2);
  __bf16* K2 = (__bf16*)carve((size_t)Bc * LV * Dc * 2);
  __bf16* K3 = (__bf16*)carve((size_t)Bc * LT * Dc * 2);
  __bf16* Vtv = (__bf16*)carve((size_t)Bc * 8 * 64 * KT * 2);
  __bf16* Vtt = (__bf16*)carve((size_t)Bc * 8 * 64 * KT * 2);
  __bf16* Av  = (__bf16*)carve((size_t)Bc * LV * Dc * 2);
  __bf16* At  = (__bf16*)carve((size_t)Bc * LT * Dc * 2);
  float*  Yv  = (float*)carve((size_t)Bc * LV * Dc * 4);
  float*  Yt  = (float*)carve((size_t)Bc * LT * Dc * 4);

  // ---- convert activations + weights to bf16 ---------------------------
  {
    int nv = Bc * LV * Dc, nu = Bc * LT * Dc;
    cvt_bf16_k<<<(nv + 255) / 256, 256, 0, stream>>>(vid_feat, vid_bf, nv);
    cvt_bf16_k<<<(nu + 255) / 256, 256, 0, stream>>>(usr_feat, usr_bf, nu);
    const float* srcs[14] = {Wq, Wq + DD, Wq + 2 * DD, Wq + 3 * DD,
                             Wk, Wk + DD, Wk + 2 * DD, Wk + 3 * DD,
                             Wv, Wv + DD, Wv + 2 * DD, Wv + 3 * DD,
                             ffvW, ffuW};
    for (int i = 0; i < 14; i++)
      transpose_cvt_k<<<(int)(DD / 256), 256, 0, stream>>>(srcs[i],
                                                           Wbf + (size_t)i * DD);
  }

  // ---- projections (WMMA GEMMs) ----------------------------------------
  auto gemm = [&](const __bf16* Ap, int M, int slot, const float* bias,
                  void* outp, int mode, int Ltot, int colOff) {
    gemm_bf16_k<<<dim3(M / 16, 1, Bc), 256, 0, stream>>>(
        Ap, Wbf + (size_t)slot * DD, bias, outp, M, mode, Ltot, colOff);
  };
  // Q (mode 0, [B,H,L,64])
  gemm(vid_bf, LV, 0, bq + 0 * Dc, Q0, 0, LV, 0);
  gemm(vid_bf, LV, 1, bq + 1 * Dc, Q1, 0, LV, 0);
  gemm(usr_bf, LT, 2, bq + 2 * Dc, Q2, 0, LT, 0);
  gemm(usr_bf, LT, 3, bq + 3 * Dc, Q3, 0, LT, 0);
  // K (mode 0)
  gemm(vid_bf, LV, 4, bk + 0 * Dc, K0, 0, LV, 0);
  gemm(usr_bf, LT, 5, bk + 1 * Dc, K1, 0, LT, 0);
  gemm(vid_bf, LV, 6, bk + 2 * Dc, K2, 0, LV, 0);
  gemm(usr_bf, LT, 7, bk + 3 * Dc, K3, 0, LT, 0);
  // V transposed, concatenated (mode 1, [B,H,64,2560])
  gemm(vid_bf, LV, 8,  bv + 0 * Dc, Vtv, 1, KT, 0);
  gemm(usr_bf, LT, 9,  bv + 1 * Dc, Vtv, 1, KT, LV);
  gemm(vid_bf, LV, 10, bv + 2 * Dc, Vtt, 1, KT, 0);
  gemm(usr_bf, LT, 11, bv + 3 * Dc, Vtt, 1, KT, LV);

  // ---- fused flash attention -------------------------------------------
  attn_fused_k<<<dim3(LV / 16, 8, Bc), 32, 0, stream>>>(
      Q0, Q1, K0, K1, Vtv, vmask, vmask, umask, Av, LV, LV, LT);
  attn_fused_k<<<dim3(LT / 16, 8, Bc), 32, 0, stream>>>(
      Q2, Q3, K2, K3, Vtt, umask, vmask, umask, At, LT, LV, LT);

  // ---- FFN (mode 2, f32 out) -------------------------------------------
  gemm(Av, LV, 12, ffvb, Yv, 2, 0, 0);
  gemm(At, LT, 13, ffub, Yt, 2, 0, 0);

  // ---- residual + LayerNorm --------------------------------------------
  float* out_vid = (float*)d_out;
  float* out_usr = (float*)d_out + (size_t)Bc * LV * Dc;
  add_ln_k<<<Bc * LV, 256, 0, stream>>>(vid_feat, Yv, lnvg, lnvb, out_vid);
  add_ln_k<<<Bc * LT, 256, 0, stream>>>(usr_feat, Yt, lnug, lnub, out_usr);
}
